// GraphClassificationModel_86732569575643
// MI455X (gfx1250) — matrix operand order repaired and checked by reading, not compiled, not run on hardware
//
#include <hip/hip_runtime.h>
#include <stddef.h>
#include <stdint.h>

#define NN 100000
#define EE 1600000
#define HH 32
#define CC 10
#define GG 64

typedef __attribute__((ext_vector_type(16))) _Float16 v16h;
typedef __attribute__((ext_vector_type(8)))  _Float16 v8h;
typedef __attribute__((ext_vector_type(8)))  float    v8f;
typedef __attribute__((ext_vector_type(4)))  unsigned v4u;
typedef __attribute__((ext_vector_type(8)))  int      v8i;
typedef __attribute__((ext_vector_type(4)))  int      v4i;

union V16H { v16h v; v8h h[2]; };

#define WMMA_F16(A,B,Cacc) __builtin_amdgcn_wmma_f32_16x16x32_f16(false,(A),false,(B),(short)0,(Cacc),false,false)

static __device__ __forceinline__ V16H load_btile(const _Float16* base, int lane) {
    V16H r;
    r.h[0] = *(const v8h*)(base + lane * 16);
    r.h[1] = *(const v8h*)(base + lane * 16 + 8);
    return r;
}

static __device__ __forceinline__ float relu(float x) { return x > 0.0f ? x : 0.0f; }

// ------------------------------------------------------------- TDM 1-D copy
// Copy `ndwords` dwords from global `gptr` into LDS offset `lds_off` via the
// Tensor Data Mover (D# group0/group1 per CDNA5 ISA sec 8.3/8.4).
static __device__ __forceinline__ void tdm_load_1d(unsigned lds_off, const void* gptr,
                                                   unsigned ndwords) {
    unsigned long long ga = (unsigned long long)(uintptr_t)gptr;
    v4u g0;
    g0[0] = 1u;                                          // count=1, user mode
    g0[1] = lds_off;                                     // lds_addr
    g0[2] = (unsigned)(ga & 0xFFFFFFFFu);                // global_addr[31:0]
    g0[3] = (unsigned)((ga >> 32) & 0x1FFFFFFu) | (2u << 30);  // addr[56:32] | type=2
    v8i g1;
    g1[0] = 0x00020000;                                  // data_size=4B, no masks/pad
    g1[1] = (int)((ndwords & 0xFFFFu) << 16);            // tensor_dim0[15:0]
    g1[2] = (int)((ndwords >> 16) & 0xFFFFu) | (1 << 16);// tensor_dim0[31:16], tensor_dim1=1
    g1[3] = (int)((ndwords & 0xFFFFu) << 16);            // tile_dim0
    g1[4] = 1;                                           // tile_dim1=1, tile_dim2=0
    g1[5] = (int)ndwords;                                // tensor_dim0_stride[31:0]
    g1[6] = 0;                                           // stride hi, dim1_stride lo
    g1[7] = 0;
    v4i z;  z[0] = z[1] = z[2] = z[3] = 0;
#if defined(__clang_major__) && (__clang_major__ >= 23)
    v8i z8; z8[0]=z8[1]=z8[2]=z8[3]=z8[4]=z8[5]=z8[6]=z8[7]=0;
    __builtin_amdgcn_tensor_load_to_lds(g0, g1, z, z, z8, 0);
#else
    __builtin_amdgcn_tensor_load_to_lds(g0, g1, z, z, 0);
#endif
}

// ---------------------------------------------------------------- utilities
__global__ void k_zero(float* p, long long n) {
    long long i = (long long)blockIdx.x * blockDim.x + threadIdx.x;
    long long stride = (long long)gridDim.x * blockDim.x;
    for (; i < n; i += stride) p[i] = 0.0f;
}

// Pre-swizzle all f16 weight matrices into WMMA B-operand layout:
// lane l holds column n=l&15, K = ((h&8)?16:0)+((l>=16)?8:0)+(h&7) for h=0..15, at tile[l*16+h].
__global__ void k_prep_weights(const float* __restrict__ w1, const float* __restrict__ w2,
                               const float* __restrict__ g1, const float* __restrict__ g2,
                               _Float16* __restrict__ W1B, _Float16* __restrict__ W2B,
                               _Float16* __restrict__ G1B, _Float16* __restrict__ G2B) {
    for (int idx = threadIdx.x; idx < 4 * 512; idx += 256) {
        int tile = idx >> 9, rem = idx & 511, l = rem >> 4, hh = rem & 15;
        int kc = tile >> 1, nt = tile & 1;
        int k = ((hh & 8) << 1) + ((l >= 16) ? 8 : 0) + (hh & 7), n = l & 15;
        W1B[idx] = (_Float16)w1[(kc * 32 + k) * 32 + nt * 16 + n];
    }
    for (int idx = threadIdx.x; idx < 2 * 512; idx += 256) {
        int tile = idx >> 9, rem = idx & 511, l = rem >> 4, hh = rem & 15;
        int k = ((hh & 8) << 1) + ((l >= 16) ? 8 : 0) + (hh & 7), n = (l & 15) + tile * 16;
        W2B[idx] = (_Float16)w2[k * 32 + n];
    }
    for (int idx = threadIdx.x; idx < 20 * 512; idx += 256) {
        int tile = idx >> 9, rem = idx & 511, l = rem >> 4, hh = rem & 15;
        int cls = tile >> 1, nt = tile & 1;
        int k = ((hh & 8) << 1) + ((l >= 16) ? 8 : 0) + (hh & 7), n = l & 15;
        G1B[idx] = (_Float16)g1[(3 + cls * 32 + k) * 32 + nt * 16 + n];
    }
    for (int idx = threadIdx.x; idx < 2 * 512; idx += 256) {
        int tile = idx >> 9, rem = idx & 511, l = rem >> 4, hh = rem & 15;
        int k = ((hh & 8) << 1) + ((l >= 16) ? 8 : 0) + (hh & 7), n = (l & 15) + tile * 16;
        G2B[idx] = (_Float16)g2[k * 32 + n];
    }
}

// xc[c][i][j] = x[i,:] @ filt_W[c][:,j] + filt_b[c][j], stored f16
__global__ void k_xc(const float* __restrict__ x, const float* __restrict__ fW,
                     const float* __restrict__ fb, _Float16* __restrict__ XC) {
    long long tid = (long long)blockIdx.x * 256 + threadIdx.x;
    if (tid >= (long long)CC * NN * HH) return;
    int j = (int)(tid & 31);
    long long t2 = tid >> 5;
    int i = (int)(t2 % NN);
    int c = (int)(t2 / NN);
    const float* W = fW + c * 96;
    float v = fb[c * 32 + j] + x[i * 3] * W[j] + x[i * 3 + 1] * W[32 + j] + x[i * 3 + 2] * W[64 + j];
    XC[tid] = (_Float16)v;
}

__global__ void k_init_xw1(const float* __restrict__ x, const float* __restrict__ g1,
                           float* __restrict__ XW1) {
    int tid = blockIdx.x * 256 + threadIdx.x;   // exactly N*32 threads
    int j = tid & 31, i = tid >> 5;
    XW1[tid] = x[i * 3] * g1[j] + x[i * 3 + 1] * g1[32 + j] + x[i * 3 + 2] * g1[64 + j];
}

__global__ void k_deg(const int* __restrict__ col, float* __restrict__ deg) {
    int e = blockIdx.x * 256 + threadIdx.x;     // exactly E threads
    atomicAdd(&deg[col[e]], 1.0f);
}

__global__ void k_dinv(const float* __restrict__ deg, float* __restrict__ dinv) {
    int i = blockIdx.x * 256 + threadIdx.x;
    if (i < NN) dinv[i] = rsqrtf(deg[i] + 1.0f);
}

// ------------------------------------------------- dominant kernel: edge MLP
// grid = (E/512, C), block = 128 (4 waves). Each wave: 8 tiles of 16 edges.
// Per-class weights staged to LDS once per block via TDM tensor_load_to_lds.
__global__ void __launch_bounds__(128) k_edge_mlp(
    const int* __restrict__ row, const int* __restrict__ col,
    const _Float16* __restrict__ XC,
    const _Float16* __restrict__ W1B,            // W1B(2048h) + W2B(1024h) contiguous
    const _Float16* __restrict__ G1B,
    const float* __restrict__ b1, const float* __restrict__ b2,
    float* __restrict__ XW1)
{
    __shared__ __attribute__((aligned(64))) _Float16 wlds[4096];     // staged weights
    __shared__ __attribute__((aligned(64))) _Float16 ldsH[4][512];   // 16x32 f16 per wave
    __shared__ int ldsCol[4][16];

    const int wave = threadIdx.x >> 5;
    const int lane = threadIdx.x & 31;
    const int cls  = blockIdx.y;
    const int ncol = lane & 15;                 // N-column / A-row index
    const int mrow = (lane >= 16) ? 8 : 0;      // D-row base / A K-half offset

    // Stage weights: 6144 B (W1B+W2B) and 2048 B (gcn1 class block) via TDM.
    if (threadIdx.x < 32) {
        unsigned lbase = (unsigned)(uintptr_t)wlds;
        tdm_load_1d(lbase, W1B, 1536);
        tdm_load_1d(lbase + 6144, G1B + (size_t)cls * 1024, 512);
        __builtin_amdgcn_s_wait_tensorcnt(0);
    }
    __syncthreads();

    const _Float16* W1L = wlds;            // 4 tiles
    const _Float16* W2L = wlds + 2048;     // 2 tiles
    const _Float16* G1L = wlds + 3072;     // 2 tiles (this class)

    V16H b1_00 = load_btile(W1L + 0 * 512, lane);
    V16H b1_01 = load_btile(W1L + 1 * 512, lane);
    V16H b1_10 = load_btile(W1L + 2 * 512, lane);
    V16H b1_11 = load_btile(W1L + 3 * 512, lane);
    V16H b2_0  = load_btile(W2L + 0 * 512, lane);
    V16H b2_1  = load_btile(W2L + 1 * 512, lane);
    V16H b3_0  = load_btile(G1L + 0 * 512, lane);
    V16H b3_1  = load_btile(G1L + 1 * 512, lane);

    const float bias1_0 = b1[ncol];
    const float bias1_1 = b1[16 + ncol];
    const float bias2_0 = b2[ncol];
    const float bias2_1 = b2[16 + ncol];

    const _Float16* xcC = XC + (size_t)cls * NN * 32;
    _Float16* myH = ldsH[wave];
    int* myC = ldsCol[wave];

    const int tile0 = (blockIdx.x * 4 + wave) * 8;
    for (int t = 0; t < 8; ++t) {
        const int e = (tile0 + t) * 16 + ncol;
        const int r = row[e];
        const int c = col[e];
        if (lane < 16) myC[lane] = c;
        __builtin_prefetch(row + e + 16, 0, 3);
        __builtin_prefetch(col + e + 16, 0, 3);

        // A operands: gather xc rows per documented 16-bit A layout
        const _Float16* pr = xcC + (size_t)r * 32 + mrow;
        const _Float16* pc = xcC + (size_t)c * 32 + mrow;
        V16H aR, aC;
        aR.h[0] = *(const v8h*)(pr);      aR.h[1] = *(const v8h*)(pr + 16);
        aC.h[0] = *(const v8h*)(pc);      aC.h[1] = *(const v8h*)(pc + 16);

        // Layer 1: [16,64] @ [64,32] + b1
        v8f h0, h1;
        #pragma unroll
        for (int q = 0; q < 8; ++q) { h0[q] = bias1_0; h1[q] = bias1_1; }
        h0 = WMMA_F16(aR.v, b1_00.v, h0);
        h0 = WMMA_F16(aC.v, b1_10.v, h0);
        h1 = WMMA_F16(aR.v, b1_01.v, h1);
        h1 = WMMA_F16(aC.v, b1_11.v, h1);

        // ReLU + D->A transpose through LDS (same wave: LDS is in-order)
        #pragma unroll
        for (int q = 0; q < 8; ++q) {
            myH[(mrow + q) * 32 + ncol]      = (_Float16)relu(h0[q]);
            myH[(mrow + q) * 32 + 16 + ncol] = (_Float16)relu(h1[q]);
        }
        const _Float16* hp = myH + ncol * 32 + mrow;
        V16H a2;
        a2.h[0] = *(const v8h*)(hp);
        a2.h[1] = *(const v8h*)(hp + 16);

        // Layer 2: [16,32] @ [32,32] + b2 (no ReLU)
        v8f g0, g1;
        #pragma unroll
        for (int q = 0; q < 8; ++q) { g0[q] = bias2_0; g1[q] = bias2_1; }
        g0 = WMMA_F16(a2.v, b2_0.v, g0);
        g1 = WMMA_F16(a2.v, b2_1.v, g1);

        #pragma unroll
        for (int q = 0; q < 8; ++q) {
            myH[(mrow + q) * 32 + ncol]      = (_Float16)g0[q];
            myH[(mrow + q) * 32 + 16 + ncol] = (_Float16)g1[q];
        }
        V16H a3;
        a3.h[0] = *(const v8h*)(hp);
        a3.h[1] = *(const v8h*)(hp + 16);

        // Fused projection: h @ gcn1_W class block (segment_sum commutes with matmul)
        v8f o0, o1;
        #pragma unroll
        for (int q = 0; q < 8; ++q) { o0[q] = 0.0f; o1[q] = 0.0f; }
        o0 = WMMA_F16(a3.v, b3_0.v, o0);
        o1 = WMMA_F16(a3.v, b3_1.v, o1);

        #pragma unroll
        for (int q = 0; q < 8; ++q) {
            const int cm = myC[mrow + q];
            float* dst = XW1 + (size_t)cm * 32 + ncol;
            atomicAdd(dst, o0[q]);
            atomicAdd(dst + 16, o1[q]);
        }
    }
}

// xw2 = h1 @ gcn2_W via WMMA; one 16-node tile per wave
__global__ void __launch_bounds__(128) k_xw2(const float* __restrict__ Hin,
                                             const _Float16* __restrict__ G2B,
                                             float* __restrict__ XW2) {
    const int wave = threadIdx.x >> 5, lane = threadIdx.x & 31;
    const int tile = blockIdx.x * 4 + wave;
    if (tile >= NN / 16) return;
    const int ncol = lane & 15;
    const int mrow = (lane >= 16) ? 8 : 0;
    V16H bb0 = load_btile(G2B + 0 * 512, lane);
    V16H bb1 = load_btile(G2B + 1 * 512, lane);

    const float* p = Hin + (size_t)(tile * 16 + ncol) * 32 + mrow;
    V16H a;
    #pragma unroll
    for (int q = 0; q < 8; ++q) { a.v[q] = (_Float16)p[q]; a.v[8 + q] = (_Float16)p[16 + q]; }

    v8f d0, d1;
    #pragma unroll
    for (int q = 0; q < 8; ++q) { d0[q] = 0.0f; d1[q] = 0.0f; }
    d0 = WMMA_F16(a.v, bb0.v, d0);
    d1 = WMMA_F16(a.v, bb1.v, d1);

    float* o = XW2 + (size_t)(tile * 16 + mrow) * 32;
    #pragma unroll
    for (int q = 0; q < 8; ++q) {
        o[q * 32 + ncol]      = d0[q];
        o[q * 32 + 16 + ncol] = d1[q];
    }
}

// out[col] += xw[row] * dinv[row]*dinv[col]; 8 lanes x float4 per edge
__global__ void k_gcn_agg(const int* __restrict__ row, const int* __restrict__ col,
                          const float* __restrict__ dinv,
                          const float* __restrict__ xw, float* __restrict__ out) {
    int tid = blockIdx.x * 256 + threadIdx.x;    // exactly E*8 threads
    int e = tid >> 3, q = tid & 7;
    int r = row[e], c = col[e];
    float s = dinv[r] * dinv[c];
    const float4 v = *(const float4*)(xw + (size_t)r * 32 + q * 4);
    float* o = out + (size_t)c * 32 + q * 4;
    atomicAdd(o + 0, v.x * s);
    atomicAdd(o + 1, v.y * s);
    atomicAdd(o + 2, v.z * s);
    atomicAdd(o + 3, v.w * s);
}

// out = relu(out + xw*dinv^2 + b)
__global__ void k_gcn_post(const float* __restrict__ xw, float* __restrict__ out,
                           const float* __restrict__ b, const float* __restrict__ dinv) {
    int tid = blockIdx.x * 256 + threadIdx.x;    // exactly N*32 threads
    int i = tid >> 5, j = tid & 31;
    float di = dinv[i];
    float v = out[tid] + xw[tid] * di * di + b[j];
    out[tid] = relu(v);
}

__global__ void k_pool(const float* __restrict__ h2, const int* __restrict__ batch,
                       float* __restrict__ pool, float* __restrict__ cnt) {
    int tid = blockIdx.x * 256 + threadIdx.x;    // exactly N*32 threads
    int i = tid >> 5, j = tid & 31;
    int b = batch[i];
    atomicAdd(&pool[b * 32 + j], h2[tid]);
    if (j == 0) atomicAdd(&cnt[b], 1.0f);
}

__global__ void k_cls(const float* __restrict__ pool, const float* __restrict__ cnt,
                      const float* __restrict__ W, const float* __restrict__ b,
                      float* __restrict__ out) {
    int tid = threadIdx.x;
    if (tid >= GG * CC) return;
    int g = tid / CC, k = tid % CC;
    float inv = 1.0f / fmaxf(cnt[g], 1.0f);
    float s = b[k];
    #pragma unroll
    for (int j = 0; j < 32; ++j) s += pool[g * 32 + j] * inv * W[j * CC + k];
    out[tid] = s;
}

// ------------------------------------------------------------------ launch
extern "C" void kernel_launch(void* const* d_in, const int* in_sizes, int n_in,
                              void* d_out, int out_size, void* d_ws, size_t ws_size,
                              hipStream_t stream) {
    const float* x     = (const float*)d_in[0];
    const int*   ei    = (const int*)d_in[1];
    const int*   batch = (const int*)d_in[2];
    const float* filtW = (const float*)d_in[3];
    const float* filtb = (const float*)d_in[4];
    const float* W1    = (const float*)d_in[5];
    const float* b1    = (const float*)d_in[6];
    const float* W2    = (const float*)d_in[7];
    const float* b2    = (const float*)d_in[8];
    const float* g1W   = (const float*)d_in[9];
    const float* g1b   = (const float*)d_in[10];
    const float* g2W   = (const float*)d_in[11];
    const float* g2b   = (const float*)d_in[12];
    const float* clsW  = (const float*)d_in[13];
    const float* clsb  = (const float*)d_in[14];
    float* out = (float*)d_out;
    const int* row = ei;
    const int* col = ei + EE;

    char* ws = (char*)d_ws;
    _Float16* XC   = (_Float16*)(ws + 0);              // 64,000,000 B
    float*    BUF0 = (float*)(ws + 64000000);          // xw1 / xw2
    float*    BUF1 = (float*)(ws + 76800000);          // gcn1 out -> h1
    float*    BUF2 = (float*)(ws + 89600000);          // gcn2 out -> h2
    float*    DEG  = (float*)(ws + 102400000);
    float*    DINV = (float*)(ws + 102800000);
    float*    POOL = (float*)(ws + 103200000);
    float*    CNT  = (float*)(ws + 103208192);
    _Float16* W1B  = (_Float16*)(ws + 103208448);      // + W2B contiguous at +4096 B
    _Float16* W2B  = (_Float16*)(ws + 103212544);
    _Float16* G1B  = (_Float16*)(ws + 103214592);
    _Float16* G2B  = (_Float16*)(ws + 103235072);
    if (ws_size < (size_t)103237120) return;

    k_zero<<<2048, 256, 0, stream>>>(BUF1, (long long)(2LL * NN * 32 + NN));
    k_zero<<<9, 256, 0, stream>>>(POOL, (long long)(GG * 32 + GG));

    k_prep_weights<<<1, 256, 0, stream>>>(W1, W2, g1W, g2W, W1B, W2B, G1B, G2B);
    k_xc<<<125000, 256, 0, stream>>>(x, filtW, filtb, XC);
    k_init_xw1<<<12500, 256, 0, stream>>>(x, g1W, BUF0);
    k_deg<<<6250, 256, 0, stream>>>(col, DEG);
    k_dinv<<<391, 256, 0, stream>>>(DEG, DINV);

    dim3 eg(3125, 10);
    k_edge_mlp<<<eg, 128, 0, stream>>>(row, col, XC, W1B, G1B, b1, b2, BUF0);

    k_gcn_agg<<<50000, 256, 0, stream>>>(row, col, DINV, BUF0, BUF1);
    k_gcn_post<<<12500, 256, 0, stream>>>(BUF0, BUF1, g1b, DINV);

    k_xw2<<<1563, 128, 0, stream>>>(BUF1, G2B, BUF0);
    k_gcn_agg<<<50000, 256, 0, stream>>>(row, col, DINV, BUF0, BUF2);
    k_gcn_post<<<12500, 256, 0, stream>>>(BUF0, BUF2, g2b, DINV);

    k_pool<<<12500, 256, 0, stream>>>(BUF2, batch, POOL, CNT);
    k_cls<<<1, 640, 0, stream>>>(POOL, CNT, clsW, clsb, out);
}
